// MemoryAggregationLayer_52879637348498
// MI455X (gfx1250) — compile-verified, matrix operand order checked by reading, not run on hardware
//
#include <hip/hip_runtime.h>
#include <stdint.h>

typedef __attribute__((ext_vector_type(16))) _Float16 v16h;
typedef __attribute__((ext_vector_type(8)))  float    v8f;
typedef __attribute__((ext_vector_type(4)))  int      v4i;

union HFrag { v16h h; uint32_t u[8]; };
union FAcc  { v8f  v; float    f[8]; };

constexpr int cB   = 2;
constexpr int cLQ  = 1024;
constexpr int cLKV = 4096;
constexpr int cD   = 512;
constexpr int cNH  = 8;
constexpr int cDH  = 64;
constexpr int cF   = 2048;

// gfx1250 async global->LDS path (guarded; falls back to VGPR staging)
#if defined(__has_builtin)
#if __has_builtin(__builtin_amdgcn_global_load_async_to_lds_b128) && \
    __has_builtin(__builtin_amdgcn_s_wait_asynccnt)
#define HAVE_ASYNC_LDS 1
#endif
#endif
#ifndef HAVE_ASYNC_LDS
#define HAVE_ASYNC_LDS 0
#endif

#if HAVE_ASYNC_LDS
typedef __attribute__((address_space(1))) v4i gv4i;  // global <4 x i32>
typedef __attribute__((address_space(3))) v4i lv4i;  // LDS    <4 x i32>
#endif

__device__ __forceinline__ v8f wmma_f16(const HFrag& a, const HFrag& b, v8f c) {
  // v_wmma_f32_16x16x32_f16: D = A(16x32 f16) * B(32x16 f16) + C(16x16 f32)
  return __builtin_amdgcn_wmma_f32_16x16x32_f16(false, a.h, false, b.h, (short)0, c,
                                                false, false);
}

// -------------------------------------------------------------------------
// Elementwise: out = f16(a [+ b])
// -------------------------------------------------------------------------
__global__ void addcvt_f16_kernel(const float* __restrict__ a, const float* __restrict__ b,
                                  _Float16* __restrict__ out, int n) {
  int i = blockIdx.x * 256 + threadIdx.x;
  if (i < n) {
    float v = b ? (a[i] + b[i]) : a[i];
    out[i] = (_Float16)v;
  }
}

// -------------------------------------------------------------------------
// Row LayerNorm over D=512 (+ optional positional embedding), f16 output.
// -------------------------------------------------------------------------
__global__ __launch_bounds__(256)
void ln_pe_f16_kernel(const float* __restrict__ x, const float* __restrict__ g,
                      const float* __restrict__ beta, const float* __restrict__ pe,
                      _Float16* __restrict__ out) {
  __shared__ float sm[256];
  const int row = blockIdx.x;
  const int t   = threadIdx.x;
  const float* xr = x + (size_t)row * cD;
  const float v0 = xr[t], v1 = xr[t + 256];

  sm[t] = v0 + v1;
  __syncthreads();
  for (int o = 128; o > 0; o >>= 1) { if (t < o) sm[t] += sm[t + o]; __syncthreads(); }
  const float mu = sm[0] * (1.0f / cD);
  __syncthreads();

  const float d0 = v0 - mu, d1 = v1 - mu;
  sm[t] = d0 * d0 + d1 * d1;
  __syncthreads();
  for (int o = 128; o > 0; o >>= 1) { if (t < o) sm[t] += sm[t + o]; __syncthreads(); }
  const float inv = rsqrtf(sm[0] * (1.0f / cD) + 1e-6f);

  float r0 = d0 * inv * g[t] + beta[t];
  float r1 = d1 * inv * g[t + 256] + beta[t + 256];
  if (pe) {
    r0 += pe[(size_t)row * cD + t];
    r1 += pe[(size_t)row * cD + t + 256];
  }
  out[(size_t)row * cD + t]       = (_Float16)r0;
  out[(size_t)row * cD + t + 256] = (_Float16)r1;
}

// -------------------------------------------------------------------------
// WMMA GEMM: C[M,N] = A[M,K](f16 row-major) * W[K,N](f16 row-major) + bias
// Block tile 128(M) x 64(N); 8 waves, each owns 16 rows x 64 cols (4 C tiles).
// The 32x64 weight tile is staged in LDS in B-fragment-swizzled order so each
// fragment is two contiguous ds_load_b128.
// MODE 0: f16 out; MODE 1: relu+f16 out; MODE 2: f32 residual += C.
// -------------------------------------------------------------------------
template <int MODE>
__global__ __launch_bounds__(256)
void gemm_wmma_kernel(const _Float16* __restrict__ A, const _Float16* __restrict__ W,
                      const float* __restrict__ bias, _Float16* __restrict__ Cf16,
                      float* __restrict__ resid, int M, int N, int K) {
  __shared__ _Float16 Bs[64 * 32];  // Bs[n*32 + pos(k)], fragment order per column
  const int tid  = threadIdx.x;
  const int wave = tid >> 5;
  const int lane = tid & 31;
  const int m0 = blockIdx.y * 128 + wave * 16;
  const int n0 = blockIdx.x * 64;
  const int mr = lane & 15;
  const int hf = lane >> 4;

  FAcc acc[4];
#pragma unroll
  for (int c = 0; c < 4; ++c)
#pragma unroll
    for (int r = 0; r < 8; ++r) acc[c].f[r] = 0.0f;

  for (int kk = 0; kk < K; kk += 32) {
    // Stage W[kk..kk+32)[n0..n0+64) swizzled: element (k,n) -> Bs[n*32 + pos(k)]
    // pos(k) = ((k&15)>>3)*16 + (k>=16)*8 + (k&7): per column the 16 halves a
    // lane needs are one contiguous 32B run starting at hf*16.
    for (int i = tid; i < 1024; i += 256) {
      const int k  = i >> 5;     // 0..31
      const int c2 = i & 31;     // n pair
      union { uint32_t u; _Float16 h[2]; } t;
      t.u = *(const uint32_t*)(W + (size_t)(kk + k) * N + n0 + 2 * c2);
      const int pos = ((k & 15) >> 3) * 16 + (k >> 4) * 8 + (k & 7);
      Bs[(2 * c2) * 32 + pos]     = t.h[0];
      Bs[(2 * c2 + 1) * 32 + pos] = t.h[1];
    }
    __syncthreads();

    HFrag a;
    const _Float16* arow = A + (size_t)(m0 + mr) * K + kk;
    *(uint4*)&a.u[0] = *(const uint4*)(arow + hf * 8);
    *(uint4*)&a.u[4] = *(const uint4*)(arow + 16 + hf * 8);
    __builtin_prefetch(arow + 32, 0, 3);  // next K-step (global_prefetch_b8)

#pragma unroll
    for (int c = 0; c < 4; ++c) {
      HFrag b;
      const _Float16* bp = &Bs[(c * 16 + mr) * 32 + hf * 16];
      *(uint4*)&b.u[0] = *(const uint4*)(bp);
      *(uint4*)&b.u[4] = *(const uint4*)(bp + 8);
      acc[c].v = wmma_f16(a, b, acc[c].v);
    }
    __syncthreads();
  }

#pragma unroll
  for (int c = 0; c < 4; ++c) {
    const int n  = n0 + c * 16 + mr;
    const float bv = bias[n];
#pragma unroll
    for (int r = 0; r < 8; ++r) {
      const int m = m0 + r + 8 * hf;
      float v = acc[c].f[r] + bv;
      if constexpr (MODE == 1) v = fmaxf(v, 0.0f);
      if constexpr (MODE == 2) {
        resid[(size_t)m * N + n] += v;
      } else {
        Cf16[(size_t)m * N + n] = (_Float16)v;
      }
    }
  }
}

// -------------------------------------------------------------------------
// Flash attention: O = softmax(Q K^T / 8) V per (batch, head).
// grid=(LQ/128, B*NH); block=256 (8 waves), wave owns 16 query rows.
// K staged row-major (async->LDS when available), V staged transposed so all
// B fragments are contiguous ds_load_b128. Row-sums of P computed with a WMMA
// against an all-ones B matrix (lands pre-broadcast in C layout).
// -------------------------------------------------------------------------
__global__ __launch_bounds__(256)
void attn_kernel(const _Float16* __restrict__ Qm, const _Float16* __restrict__ Km,
                 const _Float16* __restrict__ Vm, _Float16* __restrict__ Om) {
  __shared__ _Float16 Ks[32 * cDH];    // Ks[key*64 + d]
  __shared__ _Float16 Vt[cDH * 32];    // Vt[d*32 + key]   (transposed)
  __shared__ _Float16 Ps[8][16 * 32];  // per-wave P tile, P[m*32 + key]

  const int bh  = blockIdx.y;
  const int bb  = bh / cNH, hh = bh % cNH;
  const int tid = threadIdx.x, wave = tid >> 5, lane = tid & 31;
  const int m0  = blockIdx.x * 128 + wave * 16;
  const int mr  = lane & 15, hf = lane >> 4;

  // Q fragments: dh=64 contraction = 2 K-steps of 32
  HFrag aq[2];
  const _Float16* qrow = Qm + ((size_t)(bb * cLQ) + m0 + mr) * cD + hh * cDH;
#pragma unroll
  for (int s = 0; s < 2; ++s) {
    *(uint4*)&aq[s].u[0] = *(const uint4*)(qrow + s * 32 + hf * 8);
    *(uint4*)&aq[s].u[4] = *(const uint4*)(qrow + s * 32 + 16 + hf * 8);
  }

  HFrag ones;
#pragma unroll
  for (int j = 0; j < 8; ++j) ones.u[j] = 0x3C003C00u;  // f16 1.0 pairs

  float oacc[4][8];
#pragma unroll
  for (int c = 0; c < 4; ++c)
#pragma unroll
    for (int r = 0; r < 8; ++r) oacc[c][r] = 0.0f;

  float mi[8], li[8];
#pragma unroll
  for (int r = 0; r < 8; ++r) { mi[r] = -3.0e38f; li[r] = 0.0f; }

  const float scale = 0.125f;  // 1/sqrt(64)

  for (int j0 = 0; j0 < cLKV; j0 += 32) {
    const _Float16* kb = Km + ((size_t)(bb * cLKV) + j0) * cD + hh * cDH;
    const _Float16* vb = Vm + ((size_t)(bb * cLKV) + j0) * cD + hh * cDH;
    {
      const int r = tid >> 3, seg = tid & 7;  // 32 rows x 8 b128 segments
#if HAVE_ASYNC_LDS
      __builtin_amdgcn_global_load_async_to_lds_b128(
          (gv4i*)(kb + (size_t)r * cD + seg * 8),
          (lv4i*)&Ks[r * cDH + seg * 8], 0, 0);
#else
      *(uint4*)&Ks[r * cDH + seg * 8] = *(const uint4*)(kb + (size_t)r * cD + seg * 8);
#endif
      union { uint4 q; _Float16 h[8]; } tv;
      tv.q = *(const uint4*)(vb + (size_t)r * cD + seg * 8);
#pragma unroll
      for (int e = 0; e < 8; ++e) Vt[(seg * 8 + e) * 32 + r] = tv.h[e];
      __builtin_prefetch(kb + 32 * cD + (size_t)r * cD + seg * 8, 0, 3);
      __builtin_prefetch(vb + 32 * cD + (size_t)r * cD + seg * 8, 0, 3);
    }
#if HAVE_ASYNC_LDS
    __builtin_amdgcn_s_wait_asynccnt(0);
#endif
    __syncthreads();

    // S(16x32) = Q(16x64) K^T(64x32): two C tiles, 2 contraction steps each
    FAcc s0, s1;
#pragma unroll
    for (int r = 0; r < 8; ++r) { s0.f[r] = 0.0f; s1.f[r] = 0.0f; }
#pragma unroll
    for (int s = 0; s < 2; ++s) {
      HFrag b0, b1;
      const _Float16* k0p = &Ks[mr * cDH + s * 32];
      const _Float16* k1p = &Ks[(16 + mr) * cDH + s * 32];
      *(uint4*)&b0.u[0] = *(const uint4*)(k0p + hf * 8);
      *(uint4*)&b0.u[4] = *(const uint4*)(k0p + 16 + hf * 8);
      *(uint4*)&b1.u[0] = *(const uint4*)(k1p + hf * 8);
      *(uint4*)&b1.u[4] = *(const uint4*)(k1p + 16 + hf * 8);
      s0.v = wmma_f16(aq[s], b0, s0.v);
      s1.v = wmma_f16(aq[s], b1, s1.v);
    }

    // Online softmax: row max via 4 xor-shuffles; row sum via WMMA below.
    float corr[8];
#pragma unroll
    for (int r = 0; r < 8; ++r) {
      float mx = fmaxf(s0.f[r], s1.f[r]);
      mx = fmaxf(mx, __shfl_xor(mx, 1, 32));
      mx = fmaxf(mx, __shfl_xor(mx, 2, 32));
      mx = fmaxf(mx, __shfl_xor(mx, 4, 32));
      mx = fmaxf(mx, __shfl_xor(mx, 8, 32));
      mx *= scale;
      const float mnew = fmaxf(mi[r], mx);
      const float p0 = __expf(s0.f[r] * scale - mnew);
      const float p1 = __expf(s1.f[r] * scale - mnew);
      corr[r] = __expf(mi[r] - mnew);
      mi[r] = mnew;
#pragma unroll
      for (int c = 0; c < 4; ++c) oacc[c][r] *= corr[r];
      const int m = r + 8 * hf;
      Ps[wave][m * 32 + mr]      = (_Float16)p0;
      Ps[wave][m * 32 + 16 + mr] = (_Float16)p1;
    }
    __syncthreads();

    // P as A fragment (per-wave LDS region)
    HFrag ap;
    const _Float16* pr = &Ps[wave][mr * 32];
    *(uint4*)&ap.u[0] = *(const uint4*)(pr + hf * 8);
    *(uint4*)&ap.u[4] = *(const uint4*)(pr + 16 + hf * 8);

    // Row sums: rs[m][*] = P @ ones  (every lane of the half gets the sum)
    FAcc rs;
#pragma unroll
    for (int r = 0; r < 8; ++r) rs.f[r] = 0.0f;
    rs.v = wmma_f16(ap, ones, rs.v);
#pragma unroll
    for (int r = 0; r < 8; ++r) li[r] = li[r] * corr[r] + rs.f[r];

    // O(16x64) += P(16x32) V(32x64): B fragments from transposed Vt
#pragma unroll
    for (int c = 0; c < 4; ++c) {
      HFrag bv;
      const _Float16* vp = &Vt[(c * 16 + mr) * 32];
      *(uint4*)&bv.u[0] = *(const uint4*)(vp + hf * 8);
      *(uint4*)&bv.u[4] = *(const uint4*)(vp + 16 + hf * 8);
      FAcc t;
#pragma unroll
      for (int r2 = 0; r2 < 8; ++r2) t.f[r2] = oacc[c][r2];
      t.v = wmma_f16(ap, bv, t.v);
#pragma unroll
      for (int r2 = 0; r2 < 8; ++r2) oacc[c][r2] = t.f[r2];
    }
    __syncthreads();
  }

  // Epilogue: normalize and scatter to (B, Lq, NH*DH)
#pragma unroll
  for (int c = 0; c < 4; ++c)
#pragma unroll
    for (int r = 0; r < 8; ++r) {
      const int m = m0 + r + 8 * hf;
      const int dcol = hh * cDH + c * 16 + mr;
      Om[((size_t)(bb * cLQ) + m) * cD + dcol] = (_Float16)(oacc[c][r] / li[r]);
    }
}

// -------------------------------------------------------------------------
// Orchestration
// -------------------------------------------------------------------------
extern "C" void kernel_launch(void* const* d_in, const int* in_sizes, int n_in,
                              void* d_out, int out_size, void* d_ws, size_t ws_size,
                              hipStream_t stream) {
  (void)in_sizes; (void)n_in; (void)out_size; (void)ws_size;

  const float* hidden    = (const float*)d_in[0];
  const float* hidden_pe = (const float*)d_in[1];
  const float* frame     = (const float*)d_in[2];
  const float* frame_pe  = (const float*)d_in[3];
  const float* obj       = (const float*)d_in[4];
  const float* obj_pe    = (const float*)d_in[5];
  const float* ffn_W1 = (const float*)d_in[26];
  const float* ffn_b1 = (const float*)d_in[27];
  const float* ffn_W2 = (const float*)d_in[28];
  const float* ffn_b2 = (const float*)d_in[29];
  const float* n3_g   = (const float*)d_in[30];
  const float* n3_b   = (const float*)d_in[31];

  const int Mq  = cB * cLQ;    // 2048
  const int Mkv = cB * cLKV;   // 8192

  char* ws = (char*)d_ws;
  size_t off = 0;
  auto carve = [&](size_t bytes) -> void* {
    void* p = ws + off;
    off += (bytes + 255) & ~(size_t)255;
    return p;
  };

  float*    xf32   = (float*)carve((size_t)Mq * cD * sizeof(float));
  _Float16* q16    = (_Float16*)carve((size_t)Mq * cD * 2);
  _Float16* kin16  = (_Float16*)carve((size_t)Mkv * cD * 2);  // reused as FFN h1
  _Float16* vin16  = (_Float16*)carve((size_t)Mkv * cD * 2);
  _Float16* Q16    = (_Float16*)carve((size_t)Mq * cD * 2);
  _Float16* K16    = (_Float16*)carve((size_t)Mkv * cD * 2);
  _Float16* V16    = (_Float16*)carve((size_t)Mkv * cD * 2);
  _Float16* attn16 = (_Float16*)carve((size_t)Mq * cD * 2);
  _Float16* w16    = (_Float16*)carve((size_t)cD * cF * 2 * 2);

  (void)hipMemcpyAsync(xf32, hidden, (size_t)Mq * cD * sizeof(float),
                       hipMemcpyDeviceToDevice, stream);

  struct Sub {
    const float *Wq, *Wk, *Wv, *Wo, *bq, *bk, *bv, *bo, *g, *b, *mem, *mem_pe;
  };
  Sub subs[2] = {
    { (const float*)d_in[6],  (const float*)d_in[7],  (const float*)d_in[8],
      (const float*)d_in[9],  (const float*)d_in[10], (const float*)d_in[11],
      (const float*)d_in[12], (const float*)d_in[13], (const float*)d_in[14],
      (const float*)d_in[15], frame, frame_pe },
    { (const float*)d_in[16], (const float*)d_in[17], (const float*)d_in[18],
      (const float*)d_in[19], (const float*)d_in[20], (const float*)d_in[21],
      (const float*)d_in[22], (const float*)d_in[23], (const float*)d_in[24],
      (const float*)d_in[25], obj, obj_pe },
  };

  const int nkv = Mkv * cD;
  const int nw  = cD * cD;
  const dim3 gq(cD / 64, Mq / 128);    // (8, 16)
  const dim3 gk(cD / 64, Mkv / 128);   // (8, 64)
  const dim3 ga(cLQ / 128, cB * cNH);  // (8, 16)

  for (int si = 0; si < 2; ++si) {
    const Sub& S = subs[si];
    ln_pe_f16_kernel<<<Mq, 256, 0, stream>>>(xf32, S.g, S.b, hidden_pe, q16);
    addcvt_f16_kernel<<<(nkv + 255) / 256, 256, 0, stream>>>(S.mem, S.mem_pe, kin16, nkv);
    addcvt_f16_kernel<<<(nkv + 255) / 256, 256, 0, stream>>>(S.mem, nullptr, vin16, nkv);
    _Float16 *wq16 = w16, *wk16 = w16 + nw, *wv16 = w16 + 2 * nw, *wo16 = w16 + 3 * nw;
    addcvt_f16_kernel<<<(nw + 255) / 256, 256, 0, stream>>>(S.Wq, nullptr, wq16, nw);
    addcvt_f16_kernel<<<(nw + 255) / 256, 256, 0, stream>>>(S.Wk, nullptr, wk16, nw);
    addcvt_f16_kernel<<<(nw + 255) / 256, 256, 0, stream>>>(S.Wv, nullptr, wv16, nw);
    addcvt_f16_kernel<<<(nw + 255) / 256, 256, 0, stream>>>(S.Wo, nullptr, wo16, nw);
    gemm_wmma_kernel<0><<<gq, 256, 0, stream>>>(q16,   wq16, S.bq, Q16, nullptr, Mq,  cD, cD);
    gemm_wmma_kernel<0><<<gk, 256, 0, stream>>>(kin16, wk16, S.bk, K16, nullptr, Mkv, cD, cD);
    gemm_wmma_kernel<0><<<gk, 256, 0, stream>>>(vin16, wv16, S.bv, V16, nullptr, Mkv, cD, cD);
    attn_kernel<<<ga, 256, 0, stream>>>(Q16, K16, V16, attn16);
    gemm_wmma_kernel<2><<<gq, 256, 0, stream>>>(attn16, wo16, S.bo, nullptr, xf32, Mq, cD, cD);
  }

  // FFN
  ln_pe_f16_kernel<<<Mq, 256, 0, stream>>>(xf32, n3_g, n3_b, nullptr, q16);
  const int nw1 = cD * cF;
  addcvt_f16_kernel<<<(nw1 + 255) / 256, 256, 0, stream>>>(ffn_W1, nullptr, w16, nw1);
  addcvt_f16_kernel<<<(nw1 + 255) / 256, 256, 0, stream>>>(ffn_W2, nullptr, w16 + nw1, nw1);
  _Float16* h1 = kin16;  // 2048 x 2048
  const dim3 g1(cF / 64, Mq / 128);  // (32, 16)
  gemm_wmma_kernel<1><<<g1, 256, 0, stream>>>(q16, w16, ffn_b1, h1, nullptr, Mq, cF, cD);
  gemm_wmma_kernel<2><<<gq, 256, 0, stream>>>(h1, w16 + nw1, ffn_b2, nullptr, xf32, Mq, cD, cF);

  // outputs: (x, frame_states, obj_states) concatenated flat
  float* out = (float*)d_out;
  (void)hipMemcpyAsync(out, xf32, (size_t)Mq * cD * sizeof(float),
                       hipMemcpyDeviceToDevice, stream);
  (void)hipMemcpyAsync(out + (size_t)Mq * cD, frame, (size_t)Mkv * cD * sizeof(float),
                       hipMemcpyDeviceToDevice, stream);
  (void)hipMemcpyAsync(out + (size_t)Mq * cD + (size_t)Mkv * cD, obj,
                       (size_t)Mkv * cD * sizeof(float), hipMemcpyDeviceToDevice, stream);
}
